// RecurrentNet_52604759441781
// MI455X (gfx1250) — matrix operand order repaired and checked by reading, not compiled, not run on hardware
//
#include <hip/hip_runtime.h>
#include <hip/hip_bf16.h>
#include <math.h>

// ---------------- problem constants ----------------
#define BB   64
#define TT   2048
#define NIN  128
#define NH   256
#define NOUT 128

typedef __attribute__((ext_vector_type(16))) _Float16 v16h;
typedef __attribute__((ext_vector_type(8)))  float    v8f;

// ---------------- WMMA helpers (gfx1250, wave32) ----------------
// D = A(16x32 f16) * B(32x16 f16) + C(16x16 f32)
__device__ __forceinline__ v8f wmma_f16(v16h a, v16h b, v8f c) {
  return __builtin_amdgcn_wmma_f32_16x16x32_f16(
      /*neg_a=*/false, a, /*neg_b=*/false, b,
      /*c_mod=*/(short)0, c, /*reuse_a=*/false, /*reuse_b=*/false);
}

// A-fragment (16x32, f16) from row-major f16 source, leading dim `ld` (halves).
// ISA layout: lane l -> m=l&15; VGPR v holds K = 2*(v&3) + 8*(l>>4) + (v>=4?16:0).
__device__ __forceinline__ v16h frag_a_f16(const _Float16* __restrict__ base,
                                           int ld, int lane) {
  const int half = (lane >> 4) & 1;
  const _Float16* row = base + (lane & 15) * ld;
  v16h a;
#pragma unroll
  for (int v = 0; v < 8; ++v) {
    const int kk = 2 * (v & 3) + 8 * half + ((v & 4) << 2);
    a[2 * v]     = row[kk];
    a[2 * v + 1] = row[kk + 1];
  }
  return a;
}

// Same, but source is fp32 (converted on the fly).
__device__ __forceinline__ v16h frag_a_f32(const float* __restrict__ base,
                                           int ld, int lane) {
  const int half = (lane >> 4) & 1;
  const float* row = base + (lane & 15) * ld;
  v16h a;
#pragma unroll
  for (int v = 0; v < 8; ++v) {
    const int kk = 2 * (v & 3) + 8 * half + ((v & 4) << 2);
    a[2 * v]     = (_Float16)row[kk];
    a[2 * v + 1] = (_Float16)row[kk + 1];
  }
  return a;
}

// B-fragment (32x16 K x N) sourced from weight matrix stored as W[n][k]
// row-major (leading dim `ld` halves) -> B[k][n] = W[n][k].
// ISA layout: lane l -> n=l&15; element 2v+hh holds k = 2*(v&3) + 8*(l>>4) + (v>=4?16:0) + hh.
__device__ __forceinline__ v16h frag_b_wnk(const _Float16* __restrict__ base,
                                           int ld, int lane) {
  const int half = (lane >> 4) & 1;
  const _Float16* row = base + (lane & 15) * ld;
  v16h b;
#pragma unroll
  for (int v = 0; v < 8; ++v) {
    const int kk = 2 * (v & 3) + 8 * half + ((v & 4) << 2);
    b[2 * v]     = row[kk];
    b[2 * v + 1] = row[kk + 1];
  }
  return b;
}

// ---------------- kernel 0: fp32 -> f16 weight conversion ----------------
__global__ void k0_cvt_weights(const float* __restrict__ Win,
                               const float* __restrict__ Wih,
                               const float* __restrict__ Whh,
                               const float* __restrict__ Wout,
                               _Float16* __restrict__ win16,
                               _Float16* __restrict__ wih16,
                               _Float16* __restrict__ whh16,
                               _Float16* __restrict__ wout16) {
  const int i = blockIdx.x * blockDim.x + threadIdx.x;   // grid covers NH*NH
  if (i < NH * NIN)  win16[i]  = (_Float16)Win[i];
  if (i < NH * NH)  { wih16[i] = (_Float16)Wih[i]; whh16[i] = (_Float16)Whh[i]; }
  if (i < NOUT * NH) wout16[i] = (_Float16)Wout[i];
}

// ---------------- kernel 1: fused in-layer + ih projection ----------------
// x = relu(u @ Win^T + bin); pre[t*B+b][:] = x @ Wih^T + bih  (stored f16)
// One block handles 64 consecutive bt-rows. 512 threads = 16 waves.
// wave w: m_tile = w&3 (16 rows), n_group = w>>2, n_tiles = {ng, ng+4, ng+8, ng+12}.
#define XP 264   // padded LDS pitch (halves) for the x tile
__global__ __launch_bounds__(512)
void k1_inproj(const float* __restrict__ u,        // [B*T][NIN]
               const _Float16* __restrict__ win,   // [NH][NIN]
               const float* __restrict__ bin,      // [NH]
               const _Float16* __restrict__ wih,   // [NH][NH]
               const float* __restrict__ bih,      // [NH]
               _Float16* __restrict__ pre) {       // [T*B][NH] f16
  extern __shared__ _Float16 xs[];                 // [64][XP]
  const int lane = threadIdx.x & 31;
  const int w    = threadIdx.x >> 5;   // 0..15
  const int mt   = w & 3;
  const int ng   = w >> 2;             // 0..3
  const int r0   = blockIdx.x * 64;    // first bt row of this block
  const int half = lane >> 4, nn = lane & 15;

  const v8f vzero = {0.f, 0.f, 0.f, 0.f, 0.f, 0.f, 0.f, 0.f};

  // ---- GEMM1: x = relu(u @ Win^T + bin), K = 128 ----
  v8f acc[4];
#pragma unroll
  for (int j = 0; j < 4; ++j) acc[j] = vzero;
#pragma unroll
  for (int ks = 0; ks < NIN / 32; ++ks) {
    v16h a = frag_a_f32(u + (size_t)(r0 + mt * 16) * NIN + ks * 32, NIN, lane);
#pragma unroll
    for (int j = 0; j < 4; ++j) {
      const int nt = ng + 4 * j;
      v16h b = frag_b_wnk(win + (size_t)(nt * 16) * NIN + ks * 32, NIN, lane);
      acc[j] = wmma_f16(a, b, acc[j]);
    }
  }
#pragma unroll
  for (int j = 0; j < 4; ++j) {
    const int nt = ng + 4 * j;
#pragma unroll
    for (int r = 0; r < 8; ++r) {
      const int m = mt * 16 + r + 8 * half;
      const int n = nt * 16 + nn;
      xs[m * XP + n] = (_Float16)fmaxf(acc[j][r] + bin[n], 0.f);
    }
  }
  __syncthreads();

  // ---- GEMM2: pre = x @ Wih^T + bih, K = 256 ----
  v8f acc2[4];
#pragma unroll
  for (int j = 0; j < 4; ++j) acc2[j] = vzero;
#pragma unroll
  for (int ks = 0; ks < NH / 32; ++ks) {
    v16h a = frag_a_f16(xs + (mt * 16) * XP + ks * 32, XP, lane);
#pragma unroll
    for (int j = 0; j < 4; ++j) {
      const int nt = ng + 4 * j;
      v16h b = frag_b_wnk(wih + (size_t)(nt * 16) * NH + ks * 32, NH, lane);
      acc2[j] = wmma_f16(a, b, acc2[j]);
    }
  }
#pragma unroll
  for (int j = 0; j < 4; ++j) {
    const int nt = ng + 4 * j;
#pragma unroll
    for (int r = 0; r < 8; ++r) {
      const int m  = mt * 16 + r + 8 * half;
      const int bt = r0 + m;
      const int b  = bt >> 11;         // bt / T
      const int t  = bt & (TT - 1);    // bt % T
      const int n  = nt * 16 + nn;
      pre[((size_t)(t << 6) + b) * NH + n] = (_Float16)(acc2[j][r] + bih[n]);
    }
  }
}

// ---------------- kernel 2: sequential recurrence ----------------
// h_t = relu(tanh(pre_t + h_{t-1} @ Whh^T + bhh)); hidden[b][t][:] = h_t.
// Single persistent workgroup, 1024 threads = 32 waves on one WGP.
// LDS (dynamic, 203 KB of the WGP's 320 KB):
//   - Whh staged once as f16 [256][WP]   (B operands at DS bandwidth)
//   - h double-buffered as f16 [2][64][HP]
// Per step: 512 WMMAs spread 16/wave, one barrier, 32 KB pre_t from L2
// (prefetched one step ahead), fp32 hidden store to HBM.
#define HP 264   // padded pitch for h  (DW stride 132 ≡ 4 mod 64 -> conflict-free)
#define WP 264   // padded pitch for Whh
__global__ __launch_bounds__(1024)
void k2_recurrence(const _Float16* __restrict__ pre,   // [T*64][NH]
                   const _Float16* __restrict__ whh,   // [NH][NH]
                   const float* __restrict__ bhh,      // [NH]
                   float* __restrict__ hidden) {       // [B*T][NH] fp32 (d_out)
  extern __shared__ _Float16 hs[];          // [2][64][HP] then Whh [NH][WP]
  _Float16* wsm = hs + 2 * 64 * HP;
  const int lane = threadIdx.x & 31;
  const int w    = threadIdx.x >> 5;   // 0..31
  const int mt   = w & 3;
  const int nt0  = w >> 2;             // 0..7
  const int half = lane >> 4, nn = lane & 15;

  // h_0 = 0
  for (int i = threadIdx.x; i < 2 * 64 * HP; i += blockDim.x)
    hs[i] = (_Float16)0.f;

  // Stage Whh (f16, 128 KB) into LDS once: 256 rows x 512 B;
  // 4 threads/row, 8 x uint4 (128 B) each. LDS rows 16B-aligned (528 B pitch).
  {
    const int row  = threadIdx.x >> 2;   // 0..255
    const int part = threadIdx.x & 3;    // 0..3
    const uint4* src = (const uint4*)(whh + (size_t)row * NH);
    uint4* dst = (uint4*)(wsm + row * WP);
#pragma unroll
    for (int q = 0; q < 8; ++q)
      dst[part * 8 + q] = src[part * 8 + q];
  }
  __syncthreads();

  const v8f vzero = {0.f, 0.f, 0.f, 0.f, 0.f, 0.f, 0.f, 0.f};
  int cur = 0;
  for (int t = 0; t < TT; ++t) {
    // Prefetch next timestep's pre block (32 KB) under this step's WMMAs.
    if (t + 1 < TT)
      __builtin_prefetch(pre + ((size_t)((t + 1) << 6)) * NH + threadIdx.x * 16, 0, 1);

    const _Float16* hb = hs + cur * 64 * HP + (mt * 16) * HP;
    v8f acc0 = vzero, acc1 = vzero;
#pragma unroll
    for (int ks = 0; ks < NH / 32; ++ks) {
      v16h a  = frag_a_f16(hb + ks * 32, HP, lane);
      v16h b0 = frag_b_wnk(wsm + (nt0 * 16) * WP + ks * 32, WP, lane);
      v16h b1 = frag_b_wnk(wsm + (nt0 * 16 + 128) * WP + ks * 32, WP, lane);
      acc0 = wmma_f16(a, b0, acc0);
      acc1 = wmma_f16(a, b1, acc1);
    }

    const _Float16* prow = pre + ((size_t)(t << 6)) * NH;
    _Float16* hn = hs + (cur ^ 1) * 64 * HP;
#pragma unroll
    for (int r = 0; r < 8; ++r) {
      const int m  = mt * 16 + r + 8 * half;   // batch index
      const int n0 = nt0 * 16 + nn;
      const int n1 = n0 + 128;
      float v0 = tanhf(acc0[r] + (float)prow[m * NH + n0] + bhh[n0]);
      float v1 = tanhf(acc1[r] + (float)prow[m * NH + n1] + bhh[n1]);
      v0 = fmaxf(v0, 0.f);
      v1 = fmaxf(v1, 0.f);
      hidden[((size_t)m * TT + t) * NH + n0] = v0;
      hidden[((size_t)m * TT + t) * NH + n1] = v1;
      hn[m * HP + n0] = (_Float16)v0;
      hn[m * HP + n1] = (_Float16)v1;
    }
    __syncthreads();   // next buffer complete before anyone reads it
    cur ^= 1;
  }
}

// ---------------- kernel 3: output projection ----------------
// out = hidden @ Wout^T + bout.  256 threads = 8 waves; each wave one
// 16-row m-tile across all 8 n-tiles (N=128), K=256.
__global__ __launch_bounds__(256)
void k3_outproj(const float* __restrict__ hidden,   // [B*T][NH]
                const _Float16* __restrict__ wout,  // [NOUT][NH]
                const float* __restrict__ bout,     // [NOUT]
                float* __restrict__ out) {          // [B*T][NOUT]
  const int lane = threadIdx.x & 31;
  const int w    = threadIdx.x >> 5;                // 0..7
  const int mt   = blockIdx.x * 8 + w;              // 0..8191
  const int m0   = mt * 16;
  const int half = lane >> 4, nn = lane & 15;

  const v8f vzero = {0.f, 0.f, 0.f, 0.f, 0.f, 0.f, 0.f, 0.f};
  v8f acc[8];
#pragma unroll
  for (int j = 0; j < 8; ++j) acc[j] = vzero;
#pragma unroll
  for (int ks = 0; ks < NH / 32; ++ks) {
    v16h a = frag_a_f32(hidden + (size_t)m0 * NH + ks * 32, NH, lane);
#pragma unroll
    for (int j = 0; j < 8; ++j) {
      v16h b = frag_b_wnk(wout + (size_t)(j * 16) * NH + ks * 32, NH, lane);
      acc[j] = wmma_f16(a, b, acc[j]);
    }
  }
#pragma unroll
  for (int j = 0; j < 8; ++j) {
    const int n = j * 16 + nn;
#pragma unroll
    for (int r = 0; r < 8; ++r) {
      const int m = m0 + r + 8 * half;
      out[(size_t)m * NOUT + n] = acc[j][r] + bout[n];
    }
  }
}

// ---------------- host launcher ----------------
extern "C" void kernel_launch(void* const* d_in, const int* in_sizes, int n_in,
                              void* d_out, int out_size, void* d_ws, size_t ws_size,
                              hipStream_t stream) {
  const float* u    = (const float*)d_in[0];  // [B,T,NIN]
  const float* Win  = (const float*)d_in[1];
  const float* bin  = (const float*)d_in[2];
  const float* Wih  = (const float*)d_in[3];
  const float* bih  = (const float*)d_in[4];
  const float* Whh  = (const float*)d_in[5];
  const float* bhh  = (const float*)d_in[6];
  const float* Wout = (const float*)d_in[7];
  const float* bout = (const float*)d_in[8];

  float* hidden = (float*)d_out;                                   // [B,T,NH]
  float* outp   = hidden + (size_t)BB * TT * NH;                   // [B,T,NOUT]

  // workspace carve-out
  char* wsp = (char*)d_ws;
  _Float16* win16  = (_Float16*)(wsp);
  _Float16* wih16  = (_Float16*)(wsp + 65536);
  _Float16* whh16  = (_Float16*)(wsp + 65536 + 131072);
  _Float16* wout16 = (_Float16*)(wsp + 65536 + 131072 + 131072);
  _Float16* pre16  = (_Float16*)(wsp + 65536 + 131072 + 131072 + 65536); // 64 MiB

  // 0) weight conversion (covers NH*NH = 65536 elements)
  k0_cvt_weights<<<256, 256, 0, stream>>>(Win, Wih, Whh, Wout,
                                          win16, wih16, whh16, wout16);

  // 1) fused input + ih projection: (B*T)/64 = 2048 blocks
  k1_inproj<<<(BB * TT) / 64, 512, 64 * XP * sizeof(_Float16), stream>>>(
      u, win16, bin, wih16, bih, pre16);

  // 2) sequential recurrence: one persistent workgroup, 203 KB LDS
  const size_t smem2 = (size_t)(2 * 64 * HP + NH * WP) * sizeof(_Float16);
  k2_recurrence<<<1, 1024, smem2, stream>>>(pre16, whh16, bhh, hidden);

  // 3) output projection: 8192 m-tiles / 8 per block = 1024 blocks
  k3_outproj<<<1024, 256, 0, stream>>>(hidden, wout16, bout, outp);
}
